// Decoder_36524401885237
// MI455X (gfx1250) — matrix operand order repaired and checked by reading, not compiled, not run on hardware
//
#include <hip/hip_runtime.h>
#include <hip/hip_bf16.h>

typedef __attribute__((ext_vector_type(16))) __bf16 v16bf;
typedef __attribute__((ext_vector_type(2)))  __bf16 v2bf;
typedef __attribute__((ext_vector_type(8)))  float  v8f;
typedef __attribute__((ext_vector_type(8)))  int    v8i;
typedef __attribute__((ext_vector_type(4)))  int    v4i;

#define HID 64
#define LAT 64

// ---------------------------------------------------------------- helpers
// fast 1/x via v_rcp_f32 (~1 ulp, fine at bf16 working precision)
__device__ __forceinline__ float frcp(float x) { return __builtin_amdgcn_rcpf(x); }

__device__ __forceinline__ float siluf(float x) {
  return x * frcp(1.0f + __expf(-x));
}

__device__ __forceinline__ float ftanh(float x) {
#if __has_builtin(__builtin_amdgcn_tanhf)
  return __builtin_amdgcn_tanhf(x);
#else
  float e = __expf(2.0f * x);
  return 1.0f - 2.0f * frcp(e + 1.0f);
#endif
}

__device__ __forceinline__ float wrapf(float d, float L) {
  d = d >  0.5f * L ? d - L : d;
  d = d < -0.5f * L ? d + L : d;
  return d;
}

__device__ __forceinline__ v16bf ld16bf(const void* p) {
  v8i t;
  ((v4i*)&t)[0] = ((const v4i*)p)[0];
  ((v4i*)&t)[1] = ((const v4i*)p)[1];
  return __builtin_bit_cast(v16bf, t);
}

__device__ __forceinline__ unsigned packbf(float a, float b) {
  v2bf p;
  p[0] = (__bf16)a;
  p[1] = (__bf16)b;
  return __builtin_bit_cast(unsigned, p);
}

// ---------------------------------------------------------------- zero
__global__ void k_zero(float* p, int n) {
  int i = blockIdx.x * blockDim.x + threadIdx.x;
  if (i < n) p[i] = 0.0f;
}

// ---------------------------------------------------------------- weight swizzle
// Builds, in ws:  [A1sw 8192 bf16 | A2sw 4096 | ACsw 4096]  then 320 f32 consts.
// A-layout built here (16-bit A 16x32, contiguous k, same order as B): lane L holds M=L%16; slot i(0..15):
//   K = i + (L>=16 ? 16 : 0)
__global__ void k_swz(const float* ew1, const float* eb1, const float* ew2,
                      const float* eb2, const float* cw1, const float* cb1,
                      const float* cw2, __bf16* wblk, float* cst) {
  int i = blockIdx.x * blockDim.x + threadIdx.x;
  if (i < 8192) {                       // W1^T: 4 mtiles x 4 kfrags
    int slot = i & 15, lane = (i >> 4) & 31, frag = i >> 9;
    int mt = frag >> 2, kf = frag & 3;
    int M = lane & 15;
    int K = slot + ((lane >= 16) ? 16 : 0);
    wblk[i] = (__bf16)ew1[(kf * 32 + K) * HID + mt * 16 + M];
  } else if (i < 12288) {               // W2^T: 4 mtiles x 2 kfrags
    int j = i - 8192;
    int slot = j & 15, lane = (j >> 4) & 31, frag = j >> 9;
    int mt = frag >> 1, kf = frag & 1;
    int M = lane & 15;
    int K = slot + ((lane >= 16) ? 16 : 0);
    wblk[i] = (__bf16)ew2[(kf * 32 + K) * HID + mt * 16 + M];
  } else if (i < 16384) {               // c_w1^T: 4 x 2
    int j = i - 12288;
    int slot = j & 15, lane = (j >> 4) & 31, frag = j >> 9;
    int mt = frag >> 1, kf = frag & 1;
    int M = lane & 15;
    int K = slot + ((lane >= 16) ? 16 : 0);
    wblk[i] = (__bf16)cw1[(kf * 32 + K) * HID + mt * 16 + M];
  } else if (i < 16384 + 320) {         // consts: w1last | b1 | b2 | cb1 | cw2
    int j = i - 16384;
    int r = j >> 6, n = j & 63;
    float v = (r == 0) ? ew1[128 * HID + n]
            : (r == 1) ? eb1[n]
            : (r == 2) ? eb2[n]
            : (r == 3) ? cb1[n]
                       : cw2[n];
    cst[j] = v;
  }
}

// ---------------------------------------------------------------- node embedding
__global__ void k_h(const float* x, const float* z, const int* batch,
                    const float* embw, const float* embb, __bf16* hbf, int N) {
  int n = blockIdx.x;
  if (n >= N) return;
  int j = threadIdx.x;
  const float* zr = z + (unsigned)batch[n] * LAT;
  float acc = embb[j] + x[n] * embw[j];
#pragma unroll 8
  for (int k = 0; k < LAT; ++k) acc += zr[k] * embw[(k + 1) * HID + j];
  hbf[(unsigned)n * HID + j] = (__bf16)acc;
}

// ---------------------------------------------------------------- main edge kernel
__global__ void __launch_bounds__(256)
k_edge(const int* ei, const float* pos, const float* lbox, const __bf16* hbf,
       const __bf16* wblk, float* sum, float* cnt, int E) {
  __shared__ __align__(16) unsigned char smem[34048];
  {
    const v4i* src = (const v4i*)wblk;  // wblk + consts are contiguous (34048 B)
    v4i* dst = (v4i*)smem;
    for (int i = threadIdx.x; i < 34048 / 16; i += blockDim.x) dst[i] = src[i];
  }
  __syncthreads();

  const __bf16* sA1 = (const __bf16*)smem;            // 16 frags x 512 elems
  const __bf16* sA2 = (const __bf16*)(smem + 16384);  // 8 frags
  const __bf16* sAC = (const __bf16*)(smem + 24576);  // 8 frags
  const float*  sC  = (const float*)(smem + 32768);
  const float* w1l = sC;        // W1 radial row (swizzled order == natural order)
  const float* b1  = sC + 64;
  const float* b2  = sC + 128;
  const float* cb1 = sC + 192;
  const float* cw2 = sC + 256;

  const int lane = threadIdx.x & 31;
  const int h    = lane >> 4;    // half-wave
  const int e16  = lane & 15;    // edge within tile (both halves mirror)
  const int wpb  = blockDim.x >> 5;
  unsigned wave   = blockIdx.x * wpb + (threadIdx.x >> 5);
  unsigned nwaves = gridDim.x * wpb;
  unsigned numTiles = ((unsigned)E + 15u) >> 4;

  const float lx = lbox[0], ly = lbox[1], lz = lbox[2];
  const char* hbase = (const char*)hbf;

  for (unsigned tile = wave; tile < numTiles; tile += nwaves) {
    unsigned edge = tile * 16u + (unsigned)e16;
    bool valid = edge < (unsigned)E;
    unsigned ec = valid ? edge : (unsigned)E - 1u;   // 32-bit indices everywhere
    unsigned row = (unsigned)ei[ec];
    unsigned col = (unsigned)ei[(unsigned)E + ec];

    const float* pr = pos + row * 3u;
    const float* pc = pos + col * 3u;
    float dx = wrapf(pr[0] - pc[0], lx);
    float dy = wrapf(pr[1] - pc[1], ly);
    float dz = wrapf(pr[2] - pc[2], lz);
    float radial = dx * dx + dy * dy + dz * dz;
    float inv = frcp(sqrtf(radial) + 1.0f);
    float dnx = dx * inv, dny = dy * inv, dnz = dz * inv;

    // B fragments of e_in^T (K = [h_row 0..63 | h_col 0..63]); radial is rank-1.
    unsigned hro = row * (HID * 2u) + (unsigned)h * 32u;  // byte offsets, 32-bit
    unsigned hco = col * (HID * 2u) + (unsigned)h * 32u;
    v16bf Bf[4];
    Bf[0] = ld16bf(hbase + hro);
    Bf[1] = ld16bf(hbase + hro + 64u);
    Bf[2] = ld16bf(hbase + hco);
    Bf[3] = ld16bf(hbase + hco + 64u);

    // ---------------- layer 1: Y1^T = W1^T * e_in^T (16 WMMAs, t-inner so
    // consecutive WMMAs hit different accumulators -> no D->C RAW chains)
    v8f acc[4] = {};
#pragma unroll
    for (int f = 0; f < 4; ++f)
#pragma unroll
      for (int t = 0; t < 4; ++t)
        acc[t] = __builtin_amdgcn_wmma_f32_16x16x32_bf16(
            false, ld16bf(sA1 + (t * 4 + f) * 512 + lane * 16),
            false, Bf[f], (short)0, acc[t], false, false);

    // bias + radial rank-1 + SiLU, pack bf16 pairs (rows M,M+1 share lane)
    unsigned pk[4][4];
#pragma unroll
    for (int t = 0; t < 4; ++t) {
      int cb = t * 16 + h * 8;
      float s[8];
#pragma unroll
      for (int v = 0; v < 8; ++v)
        s[v] = siluf(acc[t][v] + b1[cb + v] + radial * w1l[cb + v]);
#pragma unroll
      for (int j = 0; j < 4; ++j) pk[t][j] = packbf(s[2 * j], s[2 * j + 1]);
    }

    // C-layout -> B-layout: half-wave exchange (same edge, never cross-edge)
    v16bf Bn[2];
#pragma unroll
    for (int fr = 0; fr < 2; ++fr) {
      v8i ti;
#pragma unroll
      for (int j = 0; j < 4; ++j) {
        unsigned ca = __shfl_xor(pk[2 * fr][j], 16, 32);
        unsigned cb2 = __shfl_xor(pk[2 * fr + 1][j], 16, 32);
        ti[j]     = (int)((h == 0) ? pk[2 * fr][j] : cb2);
        ti[4 + j] = (int)((h == 0) ? ca : pk[2 * fr + 1][j]);
      }
      Bn[fr] = __builtin_bit_cast(v16bf, ti);
    }

    // ---------------- layer 2: m^T = W2^T * Y1^T  (8 WMMAs)
    v8f a2[4] = {};
#pragma unroll
    for (int f = 0; f < 2; ++f)
#pragma unroll
      for (int t = 0; t < 4; ++t)
        a2[t] = __builtin_amdgcn_wmma_f32_16x16x32_bf16(
            false, ld16bf(sA2 + (t * 2 + f) * 512 + lane * 16),
            false, Bn[f], (short)0, a2[t], false, false);

#pragma unroll
    for (int t = 0; t < 4; ++t) {
      int cb = t * 16 + h * 8;
      float s[8];
#pragma unroll
      for (int v = 0; v < 8; ++v) s[v] = siluf(a2[t][v] + b2[cb + v]);
#pragma unroll
      for (int j = 0; j < 4; ++j) pk[t][j] = packbf(s[2 * j], s[2 * j + 1]);
    }
    v16bf Bc[2];
#pragma unroll
    for (int fr = 0; fr < 2; ++fr) {
      v8i ti;
#pragma unroll
      for (int j = 0; j < 4; ++j) {
        unsigned ca = __shfl_xor(pk[2 * fr][j], 16, 32);
        unsigned cb2 = __shfl_xor(pk[2 * fr + 1][j], 16, 32);
        ti[j]     = (int)((h == 0) ? pk[2 * fr][j] : cb2);
        ti[4 + j] = (int)((h == 0) ? ca : pk[2 * fr + 1][j]);
      }
      Bc[fr] = __builtin_bit_cast(v16bf, ti);
    }

    // ---------------- coord MLP hidden: yc^T = c_w1^T * m^T  (8 WMMAs)
    v8f a3[4] = {};
#pragma unroll
    for (int f = 0; f < 2; ++f)
#pragma unroll
      for (int t = 0; t < 4; ++t)
        a3[t] = __builtin_amdgcn_wmma_f32_16x16x32_bf16(
            false, ld16bf(sAC + (t * 2 + f) * 512 + lane * 16),
            false, Bc[f], (short)0, a3[t], false, false);

    // phi = tanh( silu(yc) . c_w2 ) — lane holds 32 of 64 features; xor-16 reduce
    float pd = 0.0f;
#pragma unroll
    for (int t = 0; t < 4; ++t) {
      int cb = t * 16 + h * 8;
#pragma unroll
      for (int v = 0; v < 8; ++v)
        pd += siluf(a3[t][v] + cb1[cb + v]) * cw2[cb + v];
    }
    float tot = pd + __shfl_xor(pd, 16, 32);
    float phi = ftanh(tot);

    if (valid && h == 0) {
      atomicAdd(&sum[row * 3u + 0u], dnx * phi);
      atomicAdd(&sum[row * 3u + 1u], dny * phi);
      atomicAdd(&sum[row * 3u + 2u], dnz * phi);
      atomicAdd(&cnt[row], 1.0f);
    }
  }
}

// ---------------------------------------------------------------- finalize
__global__ void k_out(const float* sum, const float* cnt, const float* lbox,
                      float* out, int N) {
  int i = blockIdx.x * blockDim.x + threadIdx.x;
  if (i >= N * 3) return;
  int node = i / 3, d = i % 3;
  float c = cnt[node];
  c = c < 1.0f ? 1.0f : c;
  out[i] = wrapf(sum[i] / c, lbox[d]);   // exact IEEE divide for the final value
}

// ---------------------------------------------------------------- launch
extern "C" void kernel_launch(void* const* d_in, const int* in_sizes, int n_in,
                              void* d_out, int out_size, void* d_ws, size_t ws_size,
                              hipStream_t stream) {
  const float* x     = (const float*)d_in[0];
  const float* z     = (const float*)d_in[1];
  const float* pos   = (const float*)d_in[2];
  const int*   ei    = (const int*)d_in[3];
  const int*   batch = (const int*)d_in[4];
  const float* lbox  = (const float*)d_in[6];
  const float* embw  = (const float*)d_in[7];
  const float* embb  = (const float*)d_in[8];
  const float* ew1   = (const float*)d_in[9];
  const float* eb1   = (const float*)d_in[10];
  const float* ew2   = (const float*)d_in[11];
  const float* eb2   = (const float*)d_in[12];
  const float* cw1   = (const float*)d_in[13];
  const float* cb1   = (const float*)d_in[14];
  const float* cw2   = (const float*)d_in[15];
  int N = in_sizes[0];
  int E = in_sizes[3] / 2;

  char* w = (char*)d_ws;
  size_t off = 0;
  auto carve = [&](size_t bytes) -> char* {
    char* p = w + off;
    off = (off + bytes + 255) & ~(size_t)255;
    return p;
  };
  float*  sum  = (float*)carve((size_t)N * 3 * sizeof(float));
  float*  cnt  = (float*)carve((size_t)N * sizeof(float));
  __bf16* hbf  = (__bf16*)carve((size_t)N * HID * sizeof(__bf16));
  char*   wraw = carve(34048);                // [A1|A2|AC bf16 32768B][320 f32]
  __bf16* wblk = (__bf16*)wraw;
  float*  cst  = (float*)(wraw + 32768);

  k_zero<<<(N * 3 + 255) / 256, 256, 0, stream>>>(sum, N * 3);
  k_zero<<<(N + 255) / 256, 256, 0, stream>>>(cnt, N);
  k_swz<<<(16704 + 255) / 256, 256, 0, stream>>>(ew1, eb1, ew2, eb2, cw1, cb1,
                                                 cw2, wblk, cst);
  k_h<<<N, HID, 0, stream>>>(x, z, batch, embw, embb, hbf, N);

  int numTiles = (E + 15) / 16;
  int wpb = 8;  // 256 threads = 8 waves
  int blocks = (numTiles + wpb - 1) / wpb;
  if (blocks > 4096) blocks = 4096;
  if (blocks < 1) blocks = 1;
  k_edge<<<blocks, 256, 0, stream>>>(ei, pos, lbox, hbf, wblk, sum, cnt, E);

  k_out<<<(N * 3 + 255) / 256, 256, 0, stream>>>(sum, cnt, lbox, (float*)d_out, N);
}
